// mc_model_17867063951899
// MI455X (gfx1250) — matrix operand order repaired and checked by reading, not compiled, not run on hardware
//
#include <hip/hip_runtime.h>
#include <hip/hip_bf16.h>

typedef __attribute__((ext_vector_type(16))) _Float16 v16h;
typedef __attribute__((ext_vector_type(8)))  _Float16 v8h;
typedef __attribute__((ext_vector_type(8)))  float    v8f;

#define NT 864          // 27 waves of 32
#define NWAVE_GEMM 25   // waves 0..24 own x-columns 0..399 (16 each)
#define WAVE_C 25       // wave 25: muscle GEMM via extra "c" tile
#define WAVE_ARM 26     // wave 26: arm ODE + cart output

#define NN 400
#define BB 8
#define NMUS 6
#define KCH 13          // ceil(400/32) K-chunks, last zero-padded

struct SharedBlk {
  __align__(16) _Float16 r[16][416];   // WMMA A operand; rows 8..15 and cols 400..415 zero
  float xm[NN][9];                     // [xstars+spont | spont]
  float xstars[NN][BB];                // setup: raw xstars; scan: x state x[b][n] at [n][b]
  float cmat[NMUS][NN];                // projected readout c
  float hinp[NN];
  float cisamp[NN];                    // (5+g)/0.6968
  float M9[9][15];                     // 9x9 Gram + 6 RHS (augmented, Gauss)
  float P[NMUS][9];                    // c_prms @ xm
  float buf[10][BB][NMUS];             // muscle delay line
  float musc[BB][NMUS];                // muscle state
  float joint[BB][4];                  // arm state t1,t2,d1,d2
  float red[NT];                       // deterministic reduction scratch
};

__global__ __launch_bounds__(NT, 1)
void mc_model_kernel(const int*   __restrict__ Tp,
                     const float* __restrict__ des,
                     const float* __restrict__ W,
                     const float* __restrict__ inpw,
                     const float* __restrict__ xsp,
                     const float* __restrict__ cprm,
                     const float* __restrict__ spont,
                     const float* __restrict__ g,
                     const float* __restrict__ topo,
                     float*       __restrict__ out)
{
  __shared__ SharedBlk s;
  const int tid  = threadIdx.x;
  const int wave = tid >> 5;
  const int lane = tid & 31;

  // ================= one-time setup =================
  // xstars[n][b] = (top_obs @ xstars_prms)[n][b] + joint_targ[b] . inp_weight[n][0:2]
  for (int idx = tid; idx < NN * BB; idx += NT) {
    int n = idx >> 3, b = idx & 7;
    float acc = 0.f;
    for (int k = 0; k < NN; ++k)
      acc = fmaf(topo[n * NN + k], xsp[k * BB + b], acc);
    float jt0 = des[b * 2 + 0] - 0.25f;
    float jt1 = des[b * 2 + 1] - 0.25f;
    acc += jt0 * inpw[n * 10 + 0] + jt1 * inpw[n * 10 + 1];
    s.xstars[n][b] = acc;
  }
  __syncthreads();
  // deterministic sum of squares -> scale = sqrt(z / sumsq), z = N*B*0.2^2 = 128
  {
    float part = 0.f;
    for (int idx = tid; idx < NN * BB; idx += NT) {
      float v = s.xstars[idx >> 3][idx & 7];
      part = fmaf(v, v, part);
    }
    s.red[tid] = part;
  }
  __syncthreads();
  if (tid == 0) {
    float t = 0.f;
    for (int i = 0; i < NT; ++i) t += s.red[i];
    s.red[0] = sqrtf(128.0f / t);
  }
  __syncthreads();
  const float scale = s.red[0];
  // xm = [scale*xstars + spont | spont]; x0 overwrites xstars; r0 = relu(x0); zero padding
  for (int n = tid; n < 416; n += NT) {
    if (n < NN) {
      float sp = spont[n];
      for (int b = 0; b < BB; ++b) {
        float xv = fmaf(scale, s.xstars[n][b], sp);
        s.xm[n][b] = xv;
        s.xstars[n][b] = xv;                     // x state lives here from now on
        s.r[b][n] = (_Float16)fmaxf(xv, 0.f);
      }
      s.xm[n][8] = sp;
      s.cisamp[n] = (5.0f + g[n]) * (1.0f / 0.6968f);
      for (int b = BB; b < 16; ++b) s.r[b][n] = (_Float16)0.f;
    } else {
      for (int b = 0; b < 16; ++b) s.r[b][n] = (_Float16)0.f;
    }
  }
  for (int idx = tid; idx < 10 * BB * NMUS; idx += NT)
    ((float*)s.buf)[idx] = 0.f;
  for (int idx = tid; idx < BB * NMUS; idx += NT)
    ((float*)s.musc)[idx] = 0.f;
  if (tid < BB * 4) {                            // joint0 = [1,1,0,0]
    ((float*)s.joint)[tid] = (tid & 3) < 2 ? 1.0f : 0.0f;
  }
  __syncthreads();
  // Gram M9 = xm.T@xm (9x9), P = c_prms@xm (6x9), h_inp = spont - W@spont
  if (tid < 81) {
    int a = tid / 9, c = tid % 9;
    float acc = 0.f;
    for (int n = 0; n < NN; ++n) acc = fmaf(s.xm[n][a], s.xm[n][c], acc);
    s.M9[a][c] = acc;
  } else if (tid >= 96 && tid < 96 + NMUS * 9) {
    int t = tid - 96, m = t / 9, a = t % 9;
    float acc = 0.f;
    for (int n = 0; n < NN; ++n) acc = fmaf(cprm[m * NN + n], s.xm[n][a], acc);
    s.P[m][a] = acc;
  } else if (tid >= 160 && tid < 160 + NN) {
    int n = tid - 160;
    float acc = 0.f;
    for (int k = 0; k < NN; ++k) acc = fmaf(W[n * NN + k], spont[k], acc);
    s.hinp[n] = spont[n] - acc;
  }
  __syncthreads();
  // Solve M9 * Y = P.T (9x6) by Gauss-Jordan (SPD, no pivoting); Y -> M9[:,9:15]
  if (tid == 0) {
    for (int a = 0; a < 9; ++a)
      for (int m = 0; m < NMUS; ++m) s.M9[a][9 + m] = s.P[m][a];
    for (int p = 0; p < 9; ++p) {
      float inv = 1.0f / s.M9[p][p];
      for (int c = p; c < 15; ++c) s.M9[p][c] *= inv;
      for (int rr = 0; rr < 9; ++rr) {
        if (rr == p) continue;
        float f = s.M9[rr][p];
        for (int c = p; c < 15; ++c) s.M9[rr][c] = fmaf(-f, s.M9[p][c], s.M9[rr][c]);
      }
    }
  }
  __syncthreads();
  // c = c_prms - (Y.T applied): c[m][n] = c_prms[m][n] - sum_a Y[a][m]*xm[n][a]
  for (int idx = tid; idx < NMUS * NN; idx += NT) {
    int m = idx / NN, n = idx % NN;
    float acc = cprm[m * NN + n];
    for (int a = 0; a < 9; ++a) acc = fmaf(-s.M9[a][9 + m], s.xm[n][a], acc);
    s.cmat[m][n] = acc;
  }
  __syncthreads();

  // ========== persistent per-wave register state: ONLY the W fragments ==========
  // B-fragments of W.T (f16), resident in VGPRs for the whole scan:
  // lane L holds column n = L%16; element e holds K = kc*32 + 8*(L/16) + e (e<8) or +16+(e-8) (e>=8).
  v16h wfrag[KCH];
  if (wave <= WAVE_C) {
    #pragma unroll
    for (int kc = 0; kc < KCH; ++kc) {
      union { v16h v; _Float16 h[16]; } wf;
      int nl = lane & 15;
      int kb8 = (lane >> 4) << 3;
      #pragma unroll
      for (int e = 0; e < 16; ++e) {
        float val = 0.f;
        int k = kc * 32 + (e < 8 ? kb8 + e : 16 + kb8 + (e - 8));
        bool kv = (k < NN);
        if (wave < WAVE_C) {
          if (kv) val = W[(wave * 16 + nl) * NN + k];   // B[k][n] = W[n][k]
        } else {
          if (kv && nl < NMUS) val = s.cmat[nl][k];      // extra tile: c
        }
        wf.h[e] = (_Float16)val;
      }
      wfrag[kc] = wf.v;
    }
  }
  const int T = *Tp;
  __syncthreads();

  // ================= sequential scan =================
  for (int i = 0; i < T; ++i) {
    const int slot = i % 10;
    v8f acc = {};
    if (wave <= WAVE_C) {
      // Build A fragment from LDS r (16x32 f16 per chunk), 13 resident-B WMMAs.
      const int m  = lane & 15;
      const int kb = (lane >> 4) << 3;          // lanes<16: K 0..7/16..23 ; >=16: 8..15/24..31
      const _Float16* rrow = &s.r[m][0];
      #pragma unroll
      for (int kc = 0; kc < KCH; ++kc) {
        const _Float16* rp = rrow + kc * 32 + kb;
        v8h lo = *(const v8h*)rp;               // ds_load_b128
        v8h hi = *(const v8h*)(rp + 16);        // ds_load_b128
        v16h a = __builtin_shufflevector(lo, hi, 0,1,2,3,4,5,6,7,8,9,10,11,12,13,14,15);
        acc = __builtin_amdgcn_wmma_f32_16x16x32_f16(
                  false, a, false, wfrag[kc], (short)0, acc, false, false);
      }
    } else if (wave == WAVE_ARM && lane < BB) {
      // Arm ODE: uses delayed muscle (written 10 steps ago) -> independent of this step's GEMM.
      float jt1 = s.joint[lane][0], jt2 = s.joint[lane][1];
      float jd1 = s.joint[lane][2], jd2 = s.joint[lane][3];
      float f0 = 40.f * fmaxf(s.buf[slot][lane][0], 0.f);
      float f1 = 40.f * fmaxf(s.buf[slot][lane][1], 0.f);
      float f2 = 40.f * fmaxf(s.buf[slot][lane][2], 0.f);
      float f3 = 40.f * fmaxf(s.buf[slot][lane][3], 0.f);
      float f4 = 40.f * fmaxf(s.buf[slot][lane][4], 0.f);
      float f5 = 40.f * fmaxf(s.buf[slot][lane][5], 0.f);
      float tq0 = 0.04f * f0 - 0.04f * f1 + 0.028f * f4 - 0.035f * f5;
      float tq1 = 0.025f * f2 - 0.025f * f3 + 0.035f * f4 - 0.028f * f5;
      float c2 = __cosf(jt2), s2 = __sinf(jt2);
      float M11 = 0.16f + 0.096f * c2;          // A1 + 2*A2*c2
      float M12 = 0.045f + 0.048f * c2;         // A3 + A2*c2
      const float M22 = 0.045f;                 // A3
      float h  = 0.048f * s2;
      float C1 = -h * jd2 * (2.f * jd1 + jd2);
      float C2 =  h * jd1 * jd1;
      float F1 = 0.05f * jd1 + 0.025f * jd2;
      float F2 = 0.025f * jd1 + 0.05f * jd2;
      float r1 = tq0 - C1 - F1;
      float r2 = tq1 - C2 - F2;
      float det = M11 * M22 - M12 * M12;
      float dd1 = (M22 * r1 - M12 * r2) / det;
      float dd2 = (M11 * r2 - M12 * r1) / det;
      jt1 += 0.001f * jd1;  jt2 += 0.001f * jd2;
      jd1 += 0.001f * dd1;  jd2 += 0.001f * dd2;
      s.joint[lane][0] = jt1; s.joint[lane][1] = jt2;
      s.joint[lane][2] = jd1; s.joint[lane][3] = jd2;
      float t12 = jt1 + jt2, d12 = jd1 + jd2;
      float s1 = __sinf(jt1), c1 = __cosf(jt1);
      float s12 = __sinf(t12), c12 = __cosf(t12);
      float px = 0.3f * c1 + 0.33f * c12;
      float py = 0.3f * s1 + 0.33f * s12;
      float vx = -0.3f * s1 * jd1 - 0.33f * s12 * d12;
      float vy =  0.3f * c1 * jd1 + 0.33f * c12 * d12;
      *(float4*)(out + (size_t)i * 32 + lane * 4) = make_float4(px, py, vx, vy);
    }
    __syncthreads();   // r fully consumed; buf[slot] read before overwrite
    if (wave < NWAVE_GEMM) {
      if (lane < 16) {
        // C layout: VGPR b = batch row b, lane = column n within tile.
        int n = wave * 16 + lane;
        float t  = (float)(i + 1);
        float phi = __expf(-t * (1.0f / 60.0f)) - __expf(-t * (1.0f / 6.0f));
        float add = s.hinp[n] + s.cisamp[n] * phi;
        #pragma unroll
        for (int b = 0; b < BB; ++b) {
          float xv = s.xstars[n][b];
          float xn = xv + 0.05f * (acc[b] - xv + add);        // tau_x = 0.05
          s.xstars[n][b] = xn;
          s.r[b][n] = (_Float16)fmaxf(xn, 0.f);
        }
      }
    } else if (wave == WAVE_C) {
      if (lane < NMUS) {
        #pragma unroll
        for (int b = 0; b < BB; ++b) {
          float pre = 0.2f * acc[b];                          // tau_m = 0.2
          float lr  = pre > 0.f ? pre : 0.4f * pre;           // leaky_relu(0.4)
          float mu  = lr + 0.8f * s.musc[b][lane];
          s.musc[b][lane] = mu;
          s.buf[slot][b][lane] = mu;
        }
      }
    }
    __syncthreads();   // new r / buf / x visible to next step
  }
}

extern "C" void kernel_launch(void* const* d_in, const int* in_sizes, int n_in,
                              void* d_out, int out_size, void* d_ws, size_t ws_size,
                              hipStream_t stream) {
  (void)in_sizes; (void)n_in; (void)out_size; (void)d_ws; (void)ws_size;
  const int*   Tp    = (const int*)  d_in[0];
  const float* des   = (const float*)d_in[1];
  // d_in[2] = choice (unused)
  const float* W     = (const float*)d_in[3];
  const float* inpw  = (const float*)d_in[4];
  const float* xsp   = (const float*)d_in[5];
  const float* cprm  = (const float*)d_in[6];
  const float* spont = (const float*)d_in[7];
  const float* g     = (const float*)d_in[8];
  const float* topo  = (const float*)d_in[9];
  float* out = (float*)d_out;
  hipLaunchKernelGGL(mc_model_kernel, dim3(1), dim3(NT), 0, stream,
                     Tp, des, W, inpw, xsp, cprm, spont, g, topo, out);
}
